// perceptual_network_vm_81106162418420
// MI455X (gfx1250) — matrix operand order repaired and checked by reading, not compiled, not run on hardware
//
#include <hip/hip_runtime.h>
#include <hip/hip_bf16.h>

// ---------------------------------------------------------------------------
// MI455X (gfx1250, wave32) implementation.
//
// Phase 1: featurize (von Mises pdf + rank-1 rule proj) -> combined bf16 [T,512]
// Phase 2: pack w_ih / w_hh into WMMA-B fragment-ready bf16 layouts
// Phase 3: xw = combined @ w_ih^T + (b_ih+b_hh)  via v_wmma_f32_16x16x32_bf16
// Phase 4: sequential RNN scan, ONE workgroup of 23 waves; each wave keeps its
//          12 B-fragments of w_hh^T resident in VGPRs, h ping-pongs in LDS,
//          12 WMMAs per wave per step split over 4 independent accumulators
//          (dependent-chain depth 3, not 12) -> shorter serial critical path.
// Phase 5: fc readout.
// ---------------------------------------------------------------------------

typedef __attribute__((ext_vector_type(16))) __bf16 v16bf;
typedef __attribute__((ext_vector_type(8)))  __bf16 v8bf;
typedef __attribute__((ext_vector_type(8)))  float  v8f;

#define T_STEPS 16384
#define H_DIM   361          // hidden size
#define NPAD    368          // 361 padded to 16
#define NTILES  23           // 368 / 16
#define K_IN    489          // 128 + 361 input features
#define KPAD_IN 512          // padded to 32
#define KS_IN   16           // 512 / 32 k-steps for input GEMM
#define KS_HH   12           // 384 / 32 k-steps for recurrent GEMM (covers 368)
#define PI_F    3.14159265358979f

__device__ __forceinline__ float vonmises_pdf(float grid, float mu) {
    // pdf = exp(k*cos(d)) / (2*pi*I0(k)), computed stably as
    //       exp(k*(cos(d)-1)) * sqrt(k/(2*pi)) / poly,  I0 asymptotic series.
    const float k = 1.0f / (0.1745f * 0.1745f);   // KAPPA ~ 32.84
    float d = grid - mu;
    float poly = 1.f + 1.f/(8.f*k) + 9.f/(128.f*k*k) + 225.f/(3072.f*k*k*k);
    return __expf(k * (__cosf(d) - 1.f)) * __fsqrt_rn(k / (2.f * PI_F)) / poly;
}

// ---- Phase 1: combined[T,512] bf16 = [rule_input(128) | vm(361) | pad(23)] ---
__global__ __launch_bounds__(256)
void build_combined(const float* __restrict__ angle, const float* __restrict__ rule,
                    const float* __restrict__ w_rule, const float* __restrict__ b_rule,
                    __bf16* __restrict__ combined) {
    int idx = blockIdx.x * blockDim.x + threadIdx.x;
    if (idx >= T_STEPS * KPAD_IN) return;
    int t = idx >> 9, c = idx & 511;
    float v = 0.f;
    if (c < 128) {
        v = rule[t] * w_rule[c] + b_rule[c];
    } else if (c < K_IN) {
        int i = c - 128;
        v = vonmises_pdf((float)i * (PI_F / 180.f), angle[t] * (PI_F / 180.f));
    }
    combined[idx] = (__bf16)v;
}

// ---- Phase 2: pack weights into WMMA B-operand fragment layout --------------
// B (32x16 bf16) per-lane layout: col = lane&15, K = ((lane<16)?0:8)+{e<8 ? e : 8+e}, e=0..15
// Fragment storage: frag[(tileIdx*32 + lane)*16 + e]  (two contiguous b128/lane)
__global__ __launch_bounds__(256)
void pack_b(const float* __restrict__ w, int rows /*N*/, int cols /*K*/,
            int ksteps, __bf16* __restrict__ out) {
    int idx = blockIdx.x * blockDim.x + threadIdx.x;       // over NTILES*ksteps*512
    if (idx >= NTILES * ksteps * 512) return;
    int e    = idx & 15;
    int ln   = (idx >> 4) & 31;
    int frag = idx >> 9;                 // nt*ksteps + ks
    int nt   = frag / ksteps;
    int ks   = frag % ksteps;
    int n = nt * 16 + (ln & 15);                           // output row of W
    int k = ks * 32 + ((ln < 16) ? 0 : 8) + (e < 8 ? e : 8 + e); // inner-product dim
    float v = (n < rows && k < cols) ? w[n * cols + k] : 0.f;
    out[idx] = (__bf16)v;
}

// ---- Phase 3: xw[T,368] = combined @ w_ih^T + (b_ih+b_hh) -------------------
// One wave per 16x16 tile. A (16x32 bf16): lane row = lane&15,
// K = ((lane<16)?0:8) + {e<8 ? e : 8+e}  (two b128 loads per k-step).
// Two independent accumulators halve the WMMA RAW chain.
__global__ __launch_bounds__(256)
void gemm_xw(const __bf16* __restrict__ combined, const __bf16* __restrict__ wbih,
             const float* __restrict__ b_ih, const float* __restrict__ b_hh,
             float* __restrict__ xw) {
    int ln   = threadIdx.x & 31;
    int wave = (blockIdx.x * blockDim.x + threadIdx.x) >> 5;  // 1024*23 waves
    int mt = wave / NTILES, nt = wave % NTILES;
    int kb = (ln < 16) ? 0 : 8;
    const __bf16* arow = combined + (mt * 16 + (ln & 15)) * KPAD_IN + kb;
    v8f c0 = {}, c1 = {};
#pragma unroll
    for (int ks = 0; ks < KS_IN; ++ks) {
        v8bf lo = *(const v8bf*)(arow + ks * 32);
        v8bf hi = *(const v8bf*)(arow + ks * 32 + 16);
        v16bf a;
#pragma unroll
        for (int e = 0; e < 8; ++e) { a[e] = lo[e]; a[8 + e] = hi[e]; }
        v16bf b = *(const v16bf*)(wbih + ((size_t)(nt * KS_IN + ks) * 32 + ln) * 16);
        if (ks & 1)
            c1 = __builtin_amdgcn_wmma_f32_16x16x32_bf16(false, a, false, b,
                                                         (short)0, c1, false, false);
        else
            c0 = __builtin_amdgcn_wmma_f32_16x16x32_bf16(false, a, false, b,
                                                         (short)0, c0, false, false);
    }
    int moff = mt * 16 + ((ln < 16) ? 0 : 8);
    int ncol = nt * 16 + (ln & 15);
    float bias = (ncol < H_DIM) ? b_ih[ncol] + b_hh[ncol] : 0.f;
#pragma unroll
    for (int r = 0; r < 8; ++r)
        xw[(size_t)(moff + r) * NPAD + ncol] = c0[r] + c1[r] + bias;
}

// ---- Phase 4: sequential scan. 1 block = 23 waves = 736 threads. ------------
__global__ __launch_bounds__(736)
void rnn_scan(const __bf16* __restrict__ wbhh, const float* __restrict__ xw,
              float* __restrict__ hs_out) {
    __shared__ __bf16 hbuf[2][384];          // ping-pong hidden state (bf16)
    int tid = threadIdx.x;
    int w = tid >> 5, ln = tid & 31;

    // Preload this wave's 12 recurrent-weight B fragments into VGPRs (96 VGPRs).
    v16bf Wf[KS_HH];
#pragma unroll
    for (int ks = 0; ks < KS_HH; ++ks)
        Wf[ks] = *(const v16bf*)(wbhh + ((size_t)(w * KS_HH + ks) * 32 + ln) * 16);

    for (int i = tid; i < 2 * 384; i += 736) ((__bf16*)hbuf)[i] = (__bf16)0.f;
    __syncthreads();

    const int row = ln & 15;
    const int kb  = (ln < 16) ? 0 : 8;
    const int n   = w * 16 + (ln & 15);

    for (int t = 0; t < T_STEPS; ++t) {
        int cur = t & 1, nxt = cur ^ 1;
        // Prefetch input contribution (independent of h; overlaps the WMMAs).
        float xv = (ln < 16) ? xw[(size_t)t * NPAD + n] : 0.f;

        // 4 independent accumulators: WMMA RAW-chain depth 3 instead of 12,
        // so the XDL pipeline can overlap the matvec's WMMAs each step.
        v8f acc[4] = {{}, {}, {}, {}};
#pragma unroll
        for (int ks = 0; ks < KS_HH; ++ks) {
            v16bf a;
            if (row == 0) {          // M=1 matvec: only lanes 0 and 16 carry h
                v8bf lo = *(const v8bf*)&hbuf[cur][ks * 32 + kb];
                v8bf hi = *(const v8bf*)&hbuf[cur][ks * 32 + kb + 16];
#pragma unroll
                for (int e = 0; e < 8; ++e) { a[e] = lo[e]; a[8 + e] = hi[e]; }
            } else {
#pragma unroll
                for (int e = 0; e < 16; ++e) a[e] = (__bf16)0.f;
            }
            acc[ks & 3] = __builtin_amdgcn_wmma_f32_16x16x32_bf16(
                false, a, false, Wf[ks], (short)0, acc[ks & 3], false, false);
        }
        // C row M=0 lives in VGPR0 of lanes 0..15 (N = lane).
        if (ln < 16) {
            float h = tanhf(acc[0][0] + acc[1][0] + acc[2][0] + acc[3][0] + xv);
            hbuf[nxt][n] = (__bf16)h;
            if (n < H_DIM) hs_out[(size_t)t * H_DIM + n] = h;
        }
        __syncthreads();   // one barrier/step thanks to ping-pong h buffers
    }
}

// ---- Phase 5: outcome[t] = hs[t,:] . w_fc + b_fc ----------------------------
__global__ __launch_bounds__(256)
void fc_out(const float* __restrict__ hs, const float* __restrict__ w_fc,
            const float* __restrict__ b_fc, float* __restrict__ outc) {
    int t = blockIdx.x * blockDim.x + threadIdx.x;
    if (t >= T_STEPS) return;
    float acc = b_fc[0];
#pragma unroll 4
    for (int i = 0; i < H_DIM; ++i) acc += hs[(size_t)t * H_DIM + i] * w_fc[i];
    outc[t] = acc;
}

// ---------------------------------------------------------------------------
extern "C" void kernel_launch(void* const* d_in, const int* in_sizes, int n_in,
                              void* d_out, int out_size, void* d_ws, size_t ws_size,
                              hipStream_t stream) {
    const float* angle  = (const float*)d_in[0];
    const float* rule   = (const float*)d_in[1];
    const float* w_rule = (const float*)d_in[2];
    const float* b_rule = (const float*)d_in[3];
    const float* w_ih   = (const float*)d_in[4];
    const float* w_hh   = (const float*)d_in[5];
    const float* b_ih   = (const float*)d_in[6];
    const float* b_hh   = (const float*)d_in[7];
    const float* w_fc   = (const float*)d_in[8];
    const float* b_fc   = (const float*)d_in[9];

    // d_out = [outcome_pre (T) | out/hs (T*361)] fp32
    float* outc   = (float*)d_out;
    float* hs_out = (float*)d_out + T_STEPS;

    // Workspace carve-up (all 256B-aligned chunks).
    char* ws = (char*)d_ws;
    __bf16* combined = (__bf16*)ws;                 ws += (size_t)T_STEPS * KPAD_IN * 2;
    __bf16* wbih     = (__bf16*)ws;                 ws += (size_t)NTILES * KS_IN * 512 * 2;
    __bf16* wbhh     = (__bf16*)ws;                 ws += (size_t)NTILES * KS_HH * 512 * 2;
    float*  xw       = (float*)ws;                  ws += (size_t)T_STEPS * NPAD * 4;

    build_combined<<<(T_STEPS * KPAD_IN + 255) / 256, 256, 0, stream>>>(
        angle, rule, w_rule, b_rule, combined);

    pack_b<<<(NTILES * KS_IN * 512 + 255) / 256, 256, 0, stream>>>(
        w_ih, H_DIM, K_IN, KS_IN, wbih);
    pack_b<<<(NTILES * KS_HH * 512 + 255) / 256, 256, 0, stream>>>(
        w_hh, H_DIM, H_DIM, KS_HH, wbhh);

    // 1024 M-tiles * 23 N-tiles = 23552 waves, 8 waves / block.
    gemm_xw<<<(T_STEPS / 16) * NTILES / 8, 256, 0, stream>>>(
        combined, wbih, b_ih, b_hh, xw);

    rnn_scan<<<1, 736, 0, stream>>>(wbhh, xw, hs_out);

    fc_out<<<(T_STEPS + 255) / 256, 256, 0, stream>>>(hs_out, w_fc, b_fc, outc);
}
